// HexMeshQNet_42167988912133
// MI455X (gfx1250) — matrix operand order repaired and checked by reading, not compiled, not run on hardware
//
#include <hip/hip_runtime.h>
#include <hip/hip_bf16.h>

typedef __attribute__((ext_vector_type(16))) _Float16 v16h;
typedef __attribute__((ext_vector_type(8)))  float    v8f;

#define HDIM 64
#define TILE_M 32
#define KSTEP 32
#define LDS_STRIDE 34   // 32 + 2 pad (halves) to break bank conflicts

union H2 { _Float16 h[2]; unsigned int u; };

// ---------------------------------------------------------------------------
// Generic fill
// ---------------------------------------------------------------------------
__global__ void k_fill(float* __restrict__ p, float v, long long n) {
    long long i  = (long long)blockIdx.x * blockDim.x + threadIdx.x;
    long long st = (long long)gridDim.x * blockDim.x;
    for (; i < n; i += st) p[i] = v;
}

// ---------------------------------------------------------------------------
// Degree accumulation: deg[dst[e]] += 1  (deg pre-filled with 1.0 = self loop)
// ---------------------------------------------------------------------------
__global__ void k_deg(const int* __restrict__ dst, float* __restrict__ deg, int E) {
    int e = blockIdx.x * blockDim.x + threadIdx.x;
    if (e < E) atomicAdd(&deg[dst[e]], 1.0f);
}

__global__ void k_rsqrt_inplace(float* __restrict__ d, int n) {
    int i = blockIdx.x * blockDim.x + threadIdx.x;
    if (i < n) d[i] = __frsqrt_rn(d[i]);   // deg >= 1 always
}

// ---------------------------------------------------------------------------
// WMMA GEMM: out[M x 64] = A[M x K] (f32) @ W[K x 64] (f32); f16 operands,
// f32 accumulation via v_wmma_f32_16x16x32_f16.
// Block: 256 threads = 8 wave32s; block tile 32 rows x 64 cols;
// wave w -> rowTile = (w/4)*16, colTile = (w%4)*16.
// REQUIRES: M % 32 == 0, K % 32 == 0 (true for all call sites).
// ---------------------------------------------------------------------------
template <bool ADD_BIAS, bool RELU>
__global__ __launch_bounds__(256)
void k_gemm_wmma(const float* __restrict__ A, const float* __restrict__ W,
                 const float* __restrict__ bias, float* __restrict__ out,
                 int M, int K) {
    __shared__ _Float16 As[TILE_M * LDS_STRIDE];  // [m][k], stride 34 halves
    __shared__ _Float16 Bs[HDIM  * LDS_STRIDE];   // [n][k] (W transposed)

    const int tid  = threadIdx.x;
    const int wave = tid >> 5;
    const int lane = tid & 31;
    const int half = lane >> 4;      // 0: lanes 0-15, 1: lanes 16-31
    const int l16  = lane & 15;

    const int rowBlock = blockIdx.x * TILE_M;
    const int rowBase  = rowBlock + (wave >> 2) * 16;
    const int colBase  = (wave & 3) * 16;

    v8f c = {};

    const int kSteps = K / KSTEP;
    for (int kt = 0; kt < kSteps; ++kt) {
        const int k0 = kt * KSTEP;

        // --- stage A: 32 rows x 32 k as packed f16 pairs (512 words, 2/thread)
        #pragma unroll
        for (int i = 0; i < 2; ++i) {
            int idx = tid + i * 256;          // 0..511
            int k2  = (idx & 15) * 2;         // even k
            int m   = idx >> 4;               // 0..31
            const float2 f = *(const float2*)&A[(long long)(rowBlock + m) * K + k0 + k2];
            H2 p; p.h[0] = (_Float16)f.x; p.h[1] = (_Float16)f.y;
            *(unsigned int*)&As[m * LDS_STRIDE + k2] = p.u;
        }
        // --- stage W^T: 64 cols x 32 k as packed f16 pairs (1024 words, 4/thread)
        #pragma unroll
        for (int i = 0; i < 4; ++i) {
            int idx = tid + i * 256;          // 0..1023
            int n   = idx & 63;
            int kk  = (idx >> 6) * 2;         // even k, 0..30
            H2 p;
            p.h[0] = (_Float16)W[(long long)(k0 + kk)     * HDIM + n];
            p.h[1] = (_Float16)W[(long long)(k0 + kk + 1) * HDIM + n];
            *(unsigned int*)&Bs[n * LDS_STRIDE + kk] = p.u;
        }
        __syncthreads();

        // --- A fragment (16x32 f16): VGPR v<4: k = half*8+2v; v>=4: 16+half*8+2(v-4)
        v16h a;
        const int mA = (wave >> 2) * 16 + l16;
        #pragma unroll
        for (int v = 0; v < 8; ++v) {
            int k = (v < 4) ? (half * 8 + v * 2) : (16 + half * 8 + (v - 4) * 2);
            a[2 * v]     = As[mA * LDS_STRIDE + k];
            a[2 * v + 1] = As[mA * LDS_STRIDE + k + 1];
        }
        // --- B fragment (32x16 f16): VGPR v<4: k = half*8+2v; v>=4: 16+half*8+2(v-4)
        v16h b;
        const int nB = colBase + l16;
        #pragma unroll
        for (int v = 0; v < 8; ++v) {
            int k = (v < 4) ? (half * 8 + v * 2) : (16 + half * 8 + (v - 4) * 2);
            b[2 * v]     = Bs[nB * LDS_STRIDE + k];
            b[2 * v + 1] = Bs[nB * LDS_STRIDE + k + 1];
        }

        c = __builtin_amdgcn_wmma_f32_16x16x32_f16(
                /*neg_a=*/false, a, /*neg_b=*/false, b,
                /*c_mod=*/(short)0, c, /*reuse_a=*/false, /*reuse_b=*/false);
        __syncthreads();
    }

    // --- store C: VGPR r -> M = half*8 + r, N = colBase + l16 (no guards: M%32==0)
    const int col = colBase + l16;
    float bv = 0.0f;
    if (ADD_BIAS) bv = bias[col];
    #pragma unroll
    for (int r = 0; r < 8; ++r) {
        int row = rowBase + half * 8 + r;
        float v = ADD_BIAS ? (c[r] + bv) : c[r];
        if (RELU) v = fmaxf(v, 0.0f);
        out[(long long)row * HDIM + col] = v;
    }
}

// ---------------------------------------------------------------------------
// Message passing: one thread per (edge, float4 chunk of 16).
// agg[dst] += h[src] * dinv[src]*dinv[dst]
// ---------------------------------------------------------------------------
__global__ __launch_bounds__(256)
void k_msg(const int* __restrict__ src, const int* __restrict__ dst,
           const float* __restrict__ dinv, const float* __restrict__ h,
           float* __restrict__ agg, int E) {
    long long t = (long long)blockIdx.x * blockDim.x + threadIdx.x;
    long long total = (long long)E * 16;
    if (t >= total) return;
    int e = (int)(t >> 4);
    int c = (int)(t & 15);
    int s = src[e];
    int d = dst[e];
    float nrm = dinv[s] * dinv[d];
    const float4 v = ((const float4*)h)[(long long)s * 16 + c];
    float* o = &agg[(long long)d * HDIM + c * 4];
    atomicAdd(o + 0, v.x * nrm);
    atomicAdd(o + 1, v.y * nrm);
    atomicAdd(o + 2, v.z * nrm);
    atomicAdd(o + 3, v.w * nrm);
}

// ---------------------------------------------------------------------------
// Epilogue: hout = relu(agg + h*dinv^2 + b)   (self loop fused)
// ---------------------------------------------------------------------------
__global__ __launch_bounds__(256)
void k_bias_relu_self(const float* __restrict__ agg, const float* __restrict__ h,
                      const float* __restrict__ dinv, const float* __restrict__ b,
                      float* __restrict__ hout, int N) {
    long long t = (long long)blockIdx.x * blockDim.x + threadIdx.x;
    long long total = (long long)N * HDIM;
    if (t >= total) return;
    int i = (int)(t >> 6);
    int j = (int)(t & 63);
    float di = dinv[i];
    float v  = agg[t] + h[t] * di * di + b[j];
    hout[t]  = fmaxf(v, 0.0f);
}

// ---------------------------------------------------------------------------
// Global mean pool: partial block sums -> atomicAdd into gsum[64]
// ---------------------------------------------------------------------------
__global__ __launch_bounds__(256)
void k_pool(const float* __restrict__ h, float* __restrict__ gsum, int N) {
    int j   = threadIdx.x & 63;
    int grp = threadIdx.x >> 6;           // 0..3
    float acc = 0.0f;
    for (int i = blockIdx.x * 4 + grp; i < N; i += gridDim.x * 4)
        acc += h[(long long)i * HDIM + j];
    __shared__ float red[4][HDIM];
    red[grp][j] = acc;
    __syncthreads();
    if (grp == 0) {
        float s = red[0][j] + red[1][j] + red[2][j] + red[3][j];
        atomicAdd(&gsum[j], s);
    }
}

__global__ void k_gemb(const float* __restrict__ gsum, float* __restrict__ gemb, int N) {
    int j = threadIdx.x;
    if (j < HDIM) gemb[j] = gsum[j] / fmaxf((float)N, 1.0f);
}

// ---------------------------------------------------------------------------
// Per-sheet gather-mean, writes hh[s] = [sheet_mean(64) | g_emb(64)]
// ---------------------------------------------------------------------------
__global__ __launch_bounds__(256)
void k_sheet(const float* __restrict__ h, const int* __restrict__ sidx,
             const float* __restrict__ gemb, float* __restrict__ hh, int L) {
    int s   = blockIdx.x;
    int j   = threadIdx.x & 63;
    int grp = threadIdx.x >> 6;
    float acc = 0.0f;
    for (int l = grp; l < L; l += 4) {
        int idx = sidx[(long long)s * L + l];
        acc += h[(long long)idx * HDIM + j];
    }
    __shared__ float red[4][HDIM];
    red[grp][j] = acc;
    __syncthreads();
    if (grp == 0) {
        float m = (red[0][j] + red[1][j] + red[2][j] + red[3][j]) / (float)L;
        hh[(long long)s * 128 + j]      = m;
        hh[(long long)s * 128 + 64 + j] = gemb[j];
    }
}

// ---------------------------------------------------------------------------
// Final projection: q[s] = z[s,:] @ Wm2 + bm2
// ---------------------------------------------------------------------------
__global__ void k_final(const float* __restrict__ z, const float* __restrict__ Wm2,
                        const float* __restrict__ bm2, float* __restrict__ q, int S) {
    int s = blockIdx.x * blockDim.x + threadIdx.x;
    if (s >= S) return;
    float acc = bm2[0];
    #pragma unroll
    for (int j = 0; j < HDIM; ++j) acc += z[(long long)s * HDIM + j] * Wm2[j];
    q[s] = acc;
}

// ---------------------------------------------------------------------------
extern "C" void kernel_launch(void* const* d_in, const int* in_sizes, int n_in,
                              void* d_out, int out_size, void* d_ws, size_t ws_size,
                              hipStream_t stream) {
    const float* x    = (const float*)d_in[0];
    const int*   eidx = (const int*)  d_in[1];
    const int*   sidx = (const int*)  d_in[3];
    const float* W0   = (const float*)d_in[4];
    const float* b0   = (const float*)d_in[5];
    const float* W1   = (const float*)d_in[6];
    const float* b1   = (const float*)d_in[7];
    const float* W2   = (const float*)d_in[8];
    const float* b2   = (const float*)d_in[9];
    const float* Wm1  = (const float*)d_in[10];
    const float* bm1  = (const float*)d_in[11];
    const float* Wm2  = (const float*)d_in[12];
    const float* bm2  = (const float*)d_in[13];

    const int N = in_sizes[2];            // batch is [N]
    const int F = in_sizes[0] / N;        // 128
    const int E = in_sizes[1] / 2;
    const int L = 256;
    const int S = in_sizes[3] / L;

    const int* src = eidx;        // edge_index[0]
    const int* dst = eidx + E;    // edge_index[1]

    // workspace layout (floats)
    float* ws   = (float*)d_ws;
    float* dinv = ws;                                 // N     (deg -> dinv in place)
    float* G    = dinv + N;                           // N*64  (GEMM out)
    float* Agg  = G    + (size_t)N * HDIM;            // N*64  (scatter target)
    float* Hb   = Agg  + (size_t)N * HDIM;            // N*64  (layer output)
    float* gsum = Hb   + (size_t)N * HDIM;            // 64
    float* gemb = gsum + HDIM;                        // 64
    float* hh   = gemb + HDIM;                        // S*128
    float* z    = hh   + (size_t)S * 128;             // S*64
    (void)ws_size; (void)n_in; (void)out_size;

    const long long NH = (long long)N * HDIM;

    // --- symmetric normalization
    k_fill<<<(N + 255) / 256, 256, 0, stream>>>(dinv, 1.0f, N);
    k_deg<<<(E + 255) / 256, 256, 0, stream>>>(dst, dinv, E);
    k_rsqrt_inplace<<<(N + 255) / 256, 256, 0, stream>>>(dinv, N);

    // --- 3 GCN layers
    const float* in = x;
    int K = F;
    const float* Ws[3] = {W0, W1, W2};
    const float* bs[3] = {b0, b1, b2};
    const long long msgThreads = (long long)E * 16;
    const int msgBlocks = (int)((msgThreads + 255) / 256);
    for (int l = 0; l < 3; ++l) {
        k_gemm_wmma<false, false><<<N / TILE_M, 256, 0, stream>>>(in, Ws[l], nullptr, G, N, K);
        k_fill<<<25000, 256, 0, stream>>>(Agg, 0.0f, NH);
        k_msg<<<msgBlocks, 256, 0, stream>>>(src, dst, dinv, G, Agg, E);
        k_bias_relu_self<<<(int)((NH + 255) / 256), 256, 0, stream>>>(Agg, G, dinv, bs[l], Hb, N);
        in = Hb;
        K  = HDIM;
    }

    // --- global mean pool
    k_fill<<<1, 64, 0, stream>>>(gsum, 0.0f, HDIM);
    k_pool<<<256, 256, 0, stream>>>(Hb, gsum, N);
    k_gemb<<<1, 64, 0, stream>>>(gsum, gemb, N);

    // --- sheet means + concat with global embedding
    k_sheet<<<S, 256, 0, stream>>>(Hb, sidx, gemb, hh, L);

    // --- MLP head: relu(hh @ Wm1 + bm1) @ Wm2 + bm2
    k_gemm_wmma<true, true><<<S / TILE_M, 256, 0, stream>>>(hh, Wm1, bm1, z, S, 128);
    k_final<<<(S + 255) / 256, 256, 0, stream>>>(z, Wm2, bm2, (float*)d_out, S);
}
